// NewBPSpatialConvolution_3264175145304
// MI455X (gfx1250) — matrix operand order repaired and checked by reading, not compiled, not run on hardware
//
#include <hip/hip_runtime.h>

typedef unsigned int   u32;
typedef unsigned short u16;
typedef __attribute__((ext_vector_type(4)))  u32    u32x4;
typedef __attribute__((ext_vector_type(2)))  u32    u32x2;
typedef __attribute__((ext_vector_type(4)))  float  f32x4;
typedef __attribute__((ext_vector_type(16))) __bf16 v16bf;
typedef __attribute__((ext_vector_type(8)))  float  v8f;

#define PP     128      // patch size
#define KK     15       // kernel size
#define PAD    7
#define IMG_H  142      // PP + 2*PAD
#define IMG_W  152      // row stride in elements: >=144 data cols, multiple of 8 -> 16B rows
#define COLPAD 8        // left pad of 8 (not 7) so bf16 stores/loads stay 8/16B aligned

__device__ __forceinline__ u16 f2bf(float f) {
    u32 u = __builtin_bit_cast(u32, f);
    u32 r = u + 0x7FFFu + ((u >> 16) & 1u);   // round-to-nearest-even
    return (u16)(r >> 16);
}

union BV { v16bf v; u16 s[16]; };
union AV { v16bf v; u32x4 q[2]; };

__global__ __launch_bounds__(256) void dwconv_toeplitz_wmma(
    const float* __restrict__ patches,
    const float* __restrict__ kernels,
    float* __restrict__ out)
{
    __shared__ u16   imgS[IMG_H * IMG_W];   // padded bf16 image: 43,168 B
    __shared__ float wS[KK * 16];           // flipped kernel, row stride 16

    const int g    = blockIdx.x;            // bn*C + c
    const int tid  = threadIdx.x;
    const int lane = tid & 31;              // wave32
    const int wave = tid >> 5;              // 8 waves

    // ---------- zero padded LDS image ----------
    u32* z = (u32*)imgS;
    const int nz = (IMG_H * IMG_W) / 2;
    for (int i = tid; i < nz; i += 256) z[i] = 0u;

    // ---------- stage flipped kernel: wS[ky*16+kx] = k[g][14-ky][14-kx] ----------
    if (tid < KK * KK) {
        int r = tid / KK, c = tid % KK;
        wS[(14 - r) * 16 + (14 - c)] = kernels[(size_t)g * (KK * KK) + tid];
    }
    __syncthreads();

    // ---------- f32 patch -> bf16 padded LDS image ----------
    const f32x4* p4 = (const f32x4*)(patches + (size_t)g * (PP * PP));
    #pragma unroll
    for (int it = 0; it < 16; ++it) {
        int idx = tid + it * 256;           // 0..4095 float4s
        int r   = idx >> 5;                 // 32 float4 per row
        int c   = (idx & 31) << 2;
        f32x4 v = p4[idx];
        u32x2 pk;
        pk.x = (u32)f2bf(v.x) | ((u32)f2bf(v.y) << 16);
        pk.y = (u32)f2bf(v.z) | ((u32)f2bf(v.w) << 16);
        *(u32x2*)&imgS[(r + PAD) * IMG_W + (c + COLPAD)] = pk;   // ds_store_b64
    }
    __syncthreads();

    // ---------- build banded-Toeplitz B (32x16 bf16) per ky, in B register layout ----------
    // lane 0-15: N=lane,   K=0..7,16..23; lane 16-31: N=lane-16, K=8..15,24..31
    const int N  = lane & 15;
    const int Kb = (lane >> 4) << 4;
    v16bf Bm[KK];
    #pragma unroll
    for (int ky = 0; ky < KK; ++ky) {
        BV b;
        #pragma unroll
        for (int e = 0; e < 16; ++e) {
            int k = (Kb >> 1) + e + ((e >> 3) << 3);
            int d = k - N - 1;              // COLPAD=8 shifts the band by +1
            float f = (d >= 0 && d < KK) ? wS[ky * 16 + d] : 0.0f;
            b.s[e] = f2bf(f);
        }
        Bm[ky] = b.v;
    }

    // ---------- WMMA tiles: wave w owns tile-row w; 8 tiles x 15 WMMA ----------
    const int y0    = wave * 16;
    const int arow  = lane & 15;            // A: M = lane%16
    const int acol0 = (lane >> 4) << 3;     // A: lanes>=16 start at K=8 / K=24

    #pragma unroll 1
    for (int tx = 0; tx < 8; ++tx) {
        const int x0 = tx * 16;
        v8f acc = {};
        #pragma unroll
        for (int ky = 0; ky < KK; ++ky) {
            int off = (y0 + arow + ky) * IMG_W + x0 + acol0;     // 16B aligned
            AV a;
            a.q[0] = *(const u32x4*)&imgS[off];        // VGPR0-3: K 0-7 / 8-15
            a.q[1] = *(const u32x4*)&imgS[off + 16];   // VGPR4-7: K 16-23 / 24-31
            acc = __builtin_amdgcn_wmma_f32_16x16x32_bf16(
                false, a.v, false, Bm[ky], (short)0, acc, false, false);
        }
        // D layout: lane half -> M+8, VGPR v -> M row; N = lane%16
        float* o = out + (size_t)g * (PP * PP)
                 + (size_t)(y0 + ((lane >> 4) << 3)) * PP + x0 + N;
        #pragma unroll
        for (int v = 0; v < 8; ++v)
            o[v * PP] = acc[v];             // 64B-coalesced row stores
    }
}

extern "C" void kernel_launch(void* const* d_in, const int* in_sizes, int n_in,
                              void* d_out, int out_size, void* d_ws, size_t ws_size,
                              hipStream_t stream) {
    const float* patches = (const float*)d_in[0];
    const float* kernels = (const float*)d_in[1];
    float* out = (float*)d_out;
    const int groups = in_sizes[0] / (PP * PP);   // BN*C = 3072
    dwconv_toeplitz_wmma<<<groups, 256, 0, stream>>>(patches, kernels, out);
}
